// EdgeWorldProcessorModule_52510270161468
// MI455X (gfx1250) — matrix operand order repaired and checked by reading, not compiled, not run on hardware
//
#include <hip/hip_runtime.h>

typedef __attribute__((ext_vector_type(16))) __bf16 v16bf;
typedef __attribute__((ext_vector_type(8)))  __bf16 v8bf;
typedef __attribute__((ext_vector_type(8)))  float  v8f;

// problem constants
constexpr int Dm   = 128;           // hidden dim
constexpr int K1   = 384;           // concat dim (3*D)
constexpr int MT   = 32;            // edges per tile
constexpr int THREADS = 512;        // 16 waves

// padded LDS row strides (elements) — 16B-aligned rows, bank-conflict-free
constexpr int SW1K = K1 + 8;        // 392 bf16
constexpr int SW2K = Dm + 8;        // 136 bf16
constexpr int SAK  = K1 + 8;        // 392 bf16
constexpr int SHK  = Dm + 8;        // 136 bf16
constexpr int SH2K = Dm + 4;        // 132 f32

// LDS byte offsets
constexpr int OFF_W1T = 0;                                  // 128*392*2 = 100352
constexpr int OFF_W2T = OFF_W1T + 128 * SW1K * 2;           // +34816 -> 135168
constexpr int OFF_A   = OFF_W2T + 128 * SW2K * 2;           // union with sH2
constexpr int OFF_H2  = OFF_A;                              // 32*392*2=25088 >= 32*132*4=16896
constexpr int OFF_H   = OFF_A + MT * SAK * 2;               // 160256, +8704
constexpr int SMEM_BYTES = OFF_H + MT * SHK * 2;            // 168960 bytes (165 KB)

__global__ __launch_bounds__(THREADS)
void edge_mlp_wmma_kernel(const float* __restrict__ node_attr,
                          const int*   __restrict__ eidx,   // [2*E] senders|receivers
                          const float* __restrict__ eattr,  // [E*D]
                          const float* __restrict__ W1,     // [K1*D] row-major (k,n)
                          const float* __restrict__ b1,
                          const float* __restrict__ W2,     // [D*D]
                          const float* __restrict__ b2,
                          const float* __restrict__ gamma,
                          const float* __restrict__ beta,
                          float* __restrict__ out,          // [E*D]
                          int E, int ntiles)
{
    extern __shared__ __attribute__((aligned(32))) char smem[];
    __bf16* sW1T = (__bf16*)(smem + OFF_W1T);   // [n][k] n=0..127, k=0..383
    __bf16* sW2T = (__bf16*)(smem + OFF_W2T);   // [n][k]
    __bf16* sA   = (__bf16*)(smem + OFF_A);     // [32][384] bf16 (union w/ sH2)
    float*  sH2  = (float*) (smem + OFF_H2);    // [32][128] f32
    __bf16* sH   = (__bf16*)(smem + OFF_H);     // [32][128] bf16

    const int tid  = threadIdx.x;
    const int wave = tid >> 5;           // 0..15
    const int lane = tid & 31;
    const int hi   = lane >> 4;          // half-wave select
    const int l15  = lane & 15;
    const int nt   = wave & 7;           // N-tile (8 tiles x 16 cols)
    const int mth  = wave >> 3;          // M-half (0/1)
    const int nn   = nt * 16 + l15;      // this lane's output column

    // ---- stage transposed bf16 weights once per block ----
    for (int idx = tid * 4; idx < K1 * Dm; idx += THREADS * 4) {
        const float4 w = *(const float4*)(W1 + idx);
        const int k = idx / Dm, n = idx % Dm;
        sW1T[(n + 0) * SW1K + k] = (__bf16)w.x;
        sW1T[(n + 1) * SW1K + k] = (__bf16)w.y;
        sW1T[(n + 2) * SW1K + k] = (__bf16)w.z;
        sW1T[(n + 3) * SW1K + k] = (__bf16)w.w;
    }
    for (int idx = tid * 4; idx < Dm * Dm; idx += THREADS * 4) {
        const float4 w = *(const float4*)(W2 + idx);
        const int k = idx / Dm, n = idx % Dm;
        sW2T[(n + 0) * SW2K + k] = (__bf16)w.x;
        sW2T[(n + 1) * SW2K + k] = (__bf16)w.y;
        sW2T[(n + 2) * SW2K + k] = (__bf16)w.z;
        sW2T[(n + 3) * SW2K + k] = (__bf16)w.w;
    }
    const float bias1 = b1[nn];
    const float bias2 = b2[nn];
    __syncthreads();

    // ---- persistent loop over 32-edge tiles ----
    for (int tile = blockIdx.x; tile < ntiles; tile += gridDim.x) {
        const int e0 = tile * MT;

        // gather A tile: [32 edges][sender 0:128 | receiver 128:256 | edge 256:384] -> bf16
        for (int idx = tid * 4; idx < MT * K1; idx += THREADS * 4) {
            const int r = idx / K1;
            const int c = idx % K1;                    // multiple of 4, never crosses 128/256
            int e = e0 + r; e = (e < E) ? e : (E - 1); // clamp tail
            const float* src;
            if (c < Dm)            src = node_attr + (long)eidx[e] * Dm + c;
            else if (c < 2 * Dm)   src = node_attr + (long)eidx[E + e] * Dm + (c - Dm);
            else                   src = eattr + (long)e * Dm + (c - 2 * Dm);
            const float4 v = *(const float4*)src;
            __bf16* dst = sA + r * SAK + c;
            dst[0] = (__bf16)v.x; dst[1] = (__bf16)v.y;
            dst[2] = (__bf16)v.z; dst[3] = (__bf16)v.w;
        }
        __syncthreads();

        // ---- GEMM1: [16x384] x [384x16] -> [16x16], bias + ReLU -> sH (bf16) ----
        {
            const int arow = mth * 16 + l15;
            v8f acc = {};
            #pragma unroll
            for (int ks = 0; ks < K1 / 32; ++ks) {
                const int kb = ks * 32;
                union { v16bf v; v8bf h[2]; } a, b;
                const __bf16* ap = sA + arow * SAK + kb + hi * 8;
                a.h[0] = *(const v8bf*)(ap);
                a.h[1] = *(const v8bf*)(ap + 16);
                const __bf16* bp = sW1T + nn * SW1K + kb + hi * 8;
                b.h[0] = *(const v8bf*)(bp);
                b.h[1] = *(const v8bf*)(bp + 16);
                acc = __builtin_amdgcn_wmma_f32_16x16x32_bf16(
                    false, a.v, false, b.v, (short)0, acc, false, false);
            }
            #pragma unroll
            for (int r = 0; r < 8; ++r) {
                const int m = r + hi * 8;
                float h = acc[r] + bias1;
                h = h > 0.f ? h : 0.f;
                sH[(mth * 16 + m) * SHK + nn] = (__bf16)h;
            }
        }
        __syncthreads();

        // ---- GEMM2: [16x128] x [128x16] -> [16x16] + bias -> sH2 (f32) ----
        {
            const int arow = mth * 16 + l15;
            v8f acc = {};
            #pragma unroll
            for (int ks = 0; ks < Dm / 32; ++ks) {
                const int kb = ks * 32;
                union { v16bf v; v8bf h[2]; } a, b;
                const __bf16* ap = sH + arow * SHK + kb + hi * 8;
                a.h[0] = *(const v8bf*)(ap);
                a.h[1] = *(const v8bf*)(ap + 16);
                const __bf16* bp = sW2T + nn * SW2K + kb + hi * 8;
                b.h[0] = *(const v8bf*)(bp);
                b.h[1] = *(const v8bf*)(bp + 16);
                acc = __builtin_amdgcn_wmma_f32_16x16x32_bf16(
                    false, a.v, false, b.v, (short)0, acc, false, false);
            }
            #pragma unroll
            for (int r = 0; r < 8; ++r) {
                const int m = r + hi * 8;
                sH2[(mth * 16 + m) * SH2K + nn] = acc[r] + bias2;
            }
        }
        __syncthreads();

        // ---- LayerNorm over D=128 per edge row; wave handles 2 rows, 16 lanes/row ----
        {
            const int row = wave * 2 + hi;      // 0..31
            const int sub = l15;                // 8 cols per lane
            const float* xr = sH2 + row * SH2K + sub * 8;
            float x[8];
            float s = 0.f, s2 = 0.f;
            #pragma unroll
            for (int i = 0; i < 8; ++i) { x[i] = xr[i]; s += x[i]; s2 += x[i] * x[i]; }
            #pragma unroll
            for (int m = 1; m < 16; m <<= 1) {  // reduce across the 16-lane row group
                s  += __shfl_xor(s,  m, 32);
                s2 += __shfl_xor(s2, m, 32);
            }
            const float mean = s * (1.f / 128.f);
            const float var  = s2 * (1.f / 128.f) - mean * mean;
            const float rstd = rsqrtf(var + 1e-5f);
            const int e = e0 + row;
            if (e < E) {
                float* op = out + (long)e * Dm + sub * 8;
                const float* gp = gamma + sub * 8;
                const float* bp = beta  + sub * 8;
                #pragma unroll
                for (int i = 0; i < 8; ++i)
                    op[i] = (x[i] - mean) * rstd * gp[i] + bp[i];
            }
        }
        __syncthreads();   // sA/sH2 union reused next tile
    }
}

extern "C" void kernel_launch(void* const* d_in, const int* in_sizes, int n_in,
                              void* d_out, int out_size, void* d_ws, size_t ws_size,
                              hipStream_t stream) {
    (void)n_in; (void)out_size; (void)d_ws; (void)ws_size;
    const float* node_attr = (const float*)d_in[0];
    const int*   eidx      = (const int*)  d_in[1];   // [2, E] flat
    const float* eattr     = (const float*)d_in[2];
    const float* W1        = (const float*)d_in[3];
    const float* b1        = (const float*)d_in[4];
    const float* W2        = (const float*)d_in[5];
    const float* b2        = (const float*)d_in[6];
    const float* gamma     = (const float*)d_in[7];
    const float* beta      = (const float*)d_in[8];
    float* out = (float*)d_out;

    const int E = in_sizes[1] / 2;
    const int ntiles = (E + MT - 1) / MT;
    int blocks = 2048;                       // persistent grid; weights staged once/block
    if (blocks > ntiles) blocks = ntiles;

    hipFuncSetAttribute(reinterpret_cast<const void*>(edge_mlp_wmma_kernel),
                        hipFuncAttributeMaxDynamicSharedMemorySize, SMEM_BYTES);
    edge_mlp_wmma_kernel<<<blocks, THREADS, SMEM_BYTES, stream>>>(
        node_attr, eidx, eattr, W1, b1, W2, b2, gamma, beta, out, E, ntiles);
}
